// MultiHeadAttention_51049981281147
// MI455X (gfx1250) — matrix operand order repaired and checked by reading, not compiled, not run on hardware
//
#include <hip/hip_runtime.h>
#include <stdint.h>

// ---------------------------------------------------------------------------
// MHA for MI455X (gfx1250, wave32, WMMA):
//   bf16 convert -> 3x projection GEMM (WMMA bf16, async-LDS double buffered)
//   -> fused flash attention (TDM tensor_load_to_lds double buffered, WMMA,
//   online softmax) -> output projection GEMM (fp32 out).
// ---------------------------------------------------------------------------

#define S_LEN   512
#define DMODEL  1024
#define NHEADS  16
#define DK      64
#define BATCH   8
#define MROWS   (BATCH * S_LEN)   // 4096

typedef __attribute__((ext_vector_type(16))) __bf16    v16bf;
typedef __attribute__((ext_vector_type(8)))  float     v8f;
typedef __attribute__((ext_vector_type(4)))  uint32_t  v4u;
typedef __attribute__((ext_vector_type(8)))  int       v8i;
typedef __attribute__((ext_vector_type(4)))  int       v4i;

struct Frag32B { uint4 lo, hi; };

__device__ __forceinline__ v16bf make_frag(uint4 lo, uint4 hi) {
    Frag32B f{lo, hi};
    return __builtin_bit_cast(v16bf, f);
}

__device__ __forceinline__ v8f vzero8() {
    v8f z = {0.f, 0.f, 0.f, 0.f, 0.f, 0.f, 0.f, 0.f};
    return z;
}

__device__ __forceinline__ v8f wmma_bf16(v16bf a, v16bf b, v8f c) {
    return __builtin_amdgcn_wmma_f32_16x16x32_bf16(
        false, a, false, b, (short)0, c, false, false);
}

__device__ __forceinline__ uint16_t f2bf(float f) {
    uint32_t u = __builtin_bit_cast(uint32_t, f);
    u += 0x7fffu + ((u >> 16) & 1u);       // round-to-nearest-even
    return (uint16_t)(u >> 16);
}

// Async global -> LDS, 16 bytes per lane (SADDR + 32-bit voffset form).
__device__ __forceinline__ void async_load_b128(uint32_t lds_off, uint32_t voff,
                                                const void* base) {
    asm volatile("global_load_async_to_lds_b128 %0, %1, %2"
                 :: "v"(lds_off), "v"(voff),
                    "s"((unsigned long long)(uintptr_t)base)
                 : "memory");
}

__device__ __forceinline__ void wait_async0() {
    asm volatile("s_wait_asynccnt 0x0" ::: "memory");
}

// TDM: 2-D tile load to LDS (data_size = 2 bytes), with LDS row padding.
// dims/strides/tiles in elements. pad_interval/pad_amount are D# codes.
__device__ __forceinline__ void tdm_load_2d(uint32_t lds_off, const void* gaddr,
                                            uint32_t tensor_d0, uint32_t tensor_d1,
                                            uint32_t tile_d0, uint32_t tile_d1,
                                            uint32_t stride0,
                                            uint32_t pad_interval, uint32_t pad_amount) {
    uint64_t ga = (uint64_t)(uintptr_t)gaddr;
    v4u g0;
    g0[0] = 1u;                                        // count=1, user mode
    g0[1] = lds_off;                                   // lds_addr (bytes)
    g0[2] = (uint32_t)(ga & 0xffffffffu);              // global_addr[31:0]
    g0[3] = (uint32_t)((ga >> 32) & 0x01ffffffu)       // global_addr[56:32]
          | (2u << 30);                                // type = 2 ("image")
    v8i g1;
    g1[0] = (int)((1u << 16)                           // data_size = 2 bytes
                | (1u << 20)                           // pad_enable
                | (pad_interval << 22)
                | (pad_amount << 25));
    g1[1] = (int)((tensor_d0 & 0xffffu) << 16);        // tensor_dim0[15:0]
    g1[2] = (int)((tensor_d0 >> 16) | ((tensor_d1 & 0xffffu) << 16));
    g1[3] = (int)((tensor_d1 >> 16) | (tile_d0 << 16));
    g1[4] = (int)(tile_d1 & 0xffffu);                  // tile_dim1 (tile_dim2=0)
    g1[5] = (int)stride0;                              // tensor_dim0_stride[31:0]
    g1[6] = 0;
    g1[7] = 0;
    v4i gz4 = {0, 0, 0, 0};
    v8i gz8 = {0, 0, 0, 0, 0, 0, 0, 0};
    // 6-arg form (amdgpu-toolchain / clang-23): groups 2,3 + trailing zero
    // group + cache-policy. Groups 2/3 are zero for 2-D tensors.
    __builtin_amdgcn_tensor_load_to_lds(g0, g1, gz4, gz4, gz8, 0);
}

// ---------------------------------------------------------------------------
// fp32 -> bf16 conversion, 4 elements / thread
// ---------------------------------------------------------------------------
__global__ __launch_bounds__(256)
void k_cvt_bf16x4(const float4* __restrict__ in, uint2* __restrict__ out, int n4) {
    int i = blockIdx.x * blockDim.x + threadIdx.x;
    if (i < n4) {
        float4 v = in[i];
        uint2 r;
        r.x = (uint32_t)f2bf(v.x) | ((uint32_t)f2bf(v.y) << 16);
        r.y = (uint32_t)f2bf(v.z) | ((uint32_t)f2bf(v.w) << 16);
        out[i] = r;
    }
}

// ---------------------------------------------------------------------------
// GEMM: Y[m,n] = sum_k X[m,k] * W[n,k] + bias[n]
// Async global->LDS staging, double buffered; 8 waves x (2x4) 16x16 WMMA.
// mode 0: bf16 [B,H,S,DK]; mode 1: bf16 [B,H,DK,S]; mode 2: fp32 [M,1024].
// ---------------------------------------------------------------------------
#define GS 40   // LDS row stride (bf16 elems) for 32-wide K tiles (padded)

__global__ __launch_bounds__(256)
void k_gemm_bf16(const uint16_t* __restrict__ X,
                 const uint16_t* __restrict__ W,
                 const float*    __restrict__ bias,
                 uint16_t*       __restrict__ outBf,
                 float*          __restrict__ outF,
                 int mode) {
    __shared__ __align__(16) uint16_t As[2][128 * GS];
    __shared__ __align__(16) uint16_t Bs[2][128 * GS];

    const int tid  = threadIdx.x;
    const int lane = tid & 31;
    const int w    = tid >> 5;            // 0..7
    const int wm   = w & 3;               // 0..3 (M)
    const int wn   = w >> 2;              // 0..1 (N)
    const int m0   = blockIdx.y * 128;
    const int n0   = blockIdx.x * 128;
    const int lm   = lane & 15;
    const int lh   = lane >> 4;

    const uint32_t asBase   = (uint32_t)(uintptr_t)&As[0][0];
    const uint32_t bsBase   = (uint32_t)(uintptr_t)&Bs[0][0];
    const uint32_t bufBytes = 128 * GS * 2;

    auto issueTile = [&](int k0, int buf) {
#pragma unroll
        for (int c = 0; c < 2; ++c) {
            int idx = tid + c * 256;               // 0..511
            int row = idx >> 2;
            int seg = (idx & 3) * 8;
            uint32_t ldsOff = (uint32_t)(row * GS + seg) * 2 + (uint32_t)buf * bufBytes;
            async_load_b128(asBase + ldsOff,
                            (uint32_t)(((m0 + row) * DMODEL + k0 + seg) * 2), X);
            async_load_b128(bsBase + ldsOff,
                            (uint32_t)(((n0 + row) * DMODEL + k0 + seg) * 2), W);
        }
    };

    v8f acc[2][4];
#pragma unroll
    for (int i = 0; i < 2; ++i)
#pragma unroll
        for (int j = 0; j < 4; ++j) acc[i][j] = vzero8();

    issueTile(0, 0);
    for (int k0 = 0; k0 < DMODEL; k0 += 32) {
        const int cur = (k0 >> 5) & 1;
        wait_async0();                 // this wave's tile in LDS
        __syncthreads();               // every wave's tile in LDS
        if (k0 + 32 < DMODEL) issueTile(k0 + 32, cur ^ 1);

        v16bf a[2], b[4];
#pragma unroll
        for (int i = 0; i < 2; ++i) {
            const uint16_t* p = &As[cur][(wm * 32 + i * 16 + lm) * GS];
            a[i] = make_frag(*(const uint4*)&p[lh * 8],
                             *(const uint4*)&p[16 + lh * 8]);
        }
#pragma unroll
        for (int j = 0; j < 4; ++j) {
            const uint16_t* p = &Bs[cur][(wn * 64 + j * 16 + lm) * GS];
            b[j] = make_frag(*(const uint4*)&p[lh * 8], *(const uint4*)&p[16 + lh * 8]);
        }
#pragma unroll
        for (int i = 0; i < 2; ++i)
#pragma unroll
            for (int j = 0; j < 4; ++j)
                acc[i][j] = wmma_bf16(a[i], b[j], acc[i][j]);
        __syncthreads();               // reads done before buffer reuse
    }

    // Epilogue: C element r lives at (m = r + 8*lh, n = lm)
#pragma unroll
    for (int i = 0; i < 2; ++i) {
#pragma unroll
        for (int j = 0; j < 4; ++j) {
#pragma unroll
            for (int r = 0; r < 8; ++r) {
                int m = m0 + wm * 32 + i * 16 + r + lh * 8;
                int n = n0 + wn * 64 + j * 16 + lm;
                float v = acc[i][j][r] + bias[n];
                if (mode == 2) {
                    outF[(size_t)m * DMODEL + n] = v;
                } else {
                    int bb = m >> 9, s = m & (S_LEN - 1);
                    int h = n >> 6, dk = n & (DK - 1);
                    if (mode == 0)
                        outBf[(((size_t)(bb * NHEADS + h) * S_LEN + s) * DK) + dk] = f2bf(v);
                    else
                        outBf[(((size_t)(bb * NHEADS + h) * DK + dk) * S_LEN) + s] = f2bf(v);
                }
            }
        }
    }
}

// ---------------------------------------------------------------------------
// Fused flash attention per (b,h). K/V tiles streamed by the Tensor Data
// Mover (double buffered, LDS padding done by the D# pad fields).
// ---------------------------------------------------------------------------
#define KST 72   // 64 cols + 8 pad  (rows = 32 DW -> pad_interval code 4, pad 4 DW -> code 3)
#define VST 40   // 32 cols + 8 pad  (rows = 16 DW -> pad_interval code 3, pad 4 DW -> code 3)
#define PST 40

__global__ __launch_bounds__(128)
void k_attn(const uint16_t* __restrict__ Qh,   // [B,H,S,DK]
            const uint16_t* __restrict__ Kh,   // [B,H,S,DK]
            const uint16_t* __restrict__ Vt,   // [B,H,DK,S]
            const int*      __restrict__ mask, // [B,1,S,S]
            const float*    __restrict__ relb, // [1023, NHEADS]
            uint16_t*       __restrict__ ctx)  // [B,S,DMODEL]
{
    __shared__ __align__(16) uint16_t Ks[2][32 * KST];
    __shared__ __align__(16) uint16_t Vs[2][64 * VST];
    __shared__ __align__(16) uint16_t Ps[4][16 * PST];

    const int tid  = threadIdx.x;
    const int lane = tid & 31;
    const int w    = tid >> 5;            // 0..3
    const int lm   = lane & 15;
    const int lh   = lane >> 4;

    const int bh = blockIdx.y;            // 0..127
    const int b  = bh >> 4;
    const int h  = bh & (NHEADS - 1);
    const int q0 = blockIdx.x * 64 + w * 16;

    const size_t qkBase = (size_t)bh * S_LEN * DK;
    const size_t vtBase = (size_t)bh * DK * S_LEN;

    const uint32_t ksBase = (uint32_t)(uintptr_t)&Ks[0][0];
    const uint32_t vsBase = (uint32_t)(uintptr_t)&Vs[0][0];
    const uint32_t ksBuf  = 32 * KST * 2;
    const uint32_t vsBuf  = 64 * VST * 2;

    auto issueTile = [&](int kk0, int buf) {
        if (w == 0) {
            // K tile: tensor [S=512 rows][DK=64], tile 32x64, row-major.
            tdm_load_2d(ksBase + (uint32_t)buf * ksBuf,
                        &Kh[qkBase + (size_t)kk0 * DK],
                        /*d0=*/DK, /*d1=*/S_LEN, /*tile0=*/DK, /*tile1=*/32,
                        /*stride0=*/DK, /*padI=*/4, /*padA=*/3);
            // V tile: tensor [DK=64 rows][S=512], tile 64x32.
            tdm_load_2d(vsBase + (uint32_t)buf * vsBuf,
                        &Vt[vtBase + (size_t)kk0],
                        /*d0=*/S_LEN, /*d1=*/DK, /*tile0=*/32, /*tile1=*/DK,
                        /*stride0=*/S_LEN, /*padI=*/3, /*padA=*/3);
        }
    };

    // Q fragments held in registers: dk chunks [0,32) and [32,64)
    v16bf aq[2];
    {
        const uint16_t* qrow = &Qh[qkBase + (size_t)(q0 + lm) * DK];
#pragma unroll
        for (int i = 0; i < 2; ++i) {
            int c = i * 32 + lh * 8;
            aq[i] = make_frag(*(const uint4*)&qrow[c],
                              *(const uint4*)&qrow[c + 16]);
        }
    }

    v8f acc[4];
#pragma unroll
    for (int j = 0; j < 4; ++j) acc[j] = vzero8();
    float mrow[8], lrow[8];
#pragma unroll
    for (int r = 0; r < 8; ++r) { mrow[r] = -1e30f; lrow[r] = 0.f; }

    issueTile(0, 0);
    for (int kk0 = 0; kk0 < S_LEN; kk0 += 32) {
        const int cur = (kk0 >> 5) & 1;
        if (w == 0) __builtin_amdgcn_s_wait_tensorcnt(0);
        __syncthreads();                       // tile `cur` visible to all
        if (kk0 + 32 < S_LEN) issueTile(kk0 + 32, cur ^ 1);

        // scores: 16(q) x 32(kk), two 16x16 tiles, K-depth 64 in two WMMAs
        v8f s[2];
#pragma unroll
        for (int t = 0; t < 2; ++t) {
            const uint16_t* kp = &Ks[cur][(t * 16 + lm) * KST];
            v16bf b0 = make_frag(*(const uint4*)&kp[lh * 8],
                                 *(const uint4*)&kp[16 + lh * 8]);
            v16bf b1 = make_frag(*(const uint4*)&kp[32 + lh * 8],
                                 *(const uint4*)&kp[48 + lh * 8]);
            v8f z = vzero8();
            z = wmma_bf16(aq[0], b0, z);
            z = wmma_bf16(aq[1], b1, z);
            s[t] = z;
        }

        // bias + mask + online softmax; P -> per-wave LDS staging (bf16)
        uint16_t* pst = &Ps[w][0];
#pragma unroll
        for (int r = 0; r < 8; ++r) {
            int q = q0 + r + lh * 8;
            float sv[2];
#pragma unroll
            for (int t = 0; t < 2; ++t) {
                int kk  = kk0 + t * 16 + lm;
                int idx = kk - q + (S_LEN - 1);        // in [0, 1022]
                float bv = relb[idx * NHEADS + h];
                int mk = mask[((size_t)b * S_LEN + q) * S_LEN + kk];
                sv[t] = mk ? (s[t][r] * 0.125f + bv) : -1e9f;
            }
            float rmax = fmaxf(sv[0], sv[1]);
#pragma unroll
            for (int d = 1; d < 16; d <<= 1)
                rmax = fmaxf(rmax, __shfl_xor(rmax, d, 32));
            float mnew  = fmaxf(mrow[r], rmax);
            float alpha = __expf(mrow[r] - mnew);
            float p0 = __expf(sv[0] - mnew);
            float p1 = __expf(sv[1] - mnew);
            float ps = p0 + p1;
#pragma unroll
            for (int d = 1; d < 16; d <<= 1)
                ps += __shfl_xor(ps, d, 32);
            lrow[r] = lrow[r] * alpha + ps;
            mrow[r] = mnew;
#pragma unroll
            for (int j = 0; j < 4; ++j) acc[j][r] *= alpha;
            pst[(r + lh * 8) * PST + lm]      = f2bf(p0);
            pst[(r + lh * 8) * PST + 16 + lm] = f2bf(p1);
        }
        asm volatile("" ::: "memory");  // keep wave-local LDS store->load order

        // P as A-fragment (16x32), then out += P * V
        const uint16_t* pr = &pst[lm * PST];
        v16bf ap = make_frag(*(const uint4*)&pr[lh * 8],
                             *(const uint4*)&pr[16 + lh * 8]);
#pragma unroll
        for (int j = 0; j < 4; ++j) {
            const uint16_t* vp = &Vs[cur][(j * 16 + lm) * VST];
            v16bf bv = make_frag(*(const uint4*)&vp[lh * 8], *(const uint4*)&vp[16 + lh * 8]);
            acc[j] = wmma_bf16(ap, bv, acc[j]);
        }
    }

    // Epilogue: context[b, q, h*64 + dk] (bf16) for the O-projection GEMM
#pragma unroll
    for (int j = 0; j < 4; ++j) {
#pragma unroll
        for (int r = 0; r < 8; ++r) {
            int q = q0 + r + lh * 8;
            float v = acc[j][r] / lrow[r];
            ctx[((size_t)b * S_LEN + q) * DMODEL + h * DK + j * 16 + lm] = f2bf(v);
        }
    }
}

// ---------------------------------------------------------------------------
// Host-side orchestration
// ---------------------------------------------------------------------------
extern "C" void kernel_launch(void* const* d_in, const int* in_sizes, int n_in,
                              void* d_out, int out_size, void* d_ws, size_t ws_size,
                              hipStream_t stream) {
    (void)in_sizes; (void)n_in; (void)out_size; (void)ws_size;

    const float* q_in  = (const float*)d_in[0];
    const float* k_in  = (const float*)d_in[1];
    const float* v_in  = (const float*)d_in[2];
    const int*   maskp = (const int*)  d_in[3];
    const float* w_q   = (const float*)d_in[4];
    const float* b_q   = (const float*)d_in[5];
    const float* w_k   = (const float*)d_in[6];
    const float* b_k   = (const float*)d_in[7];
    const float* w_v   = (const float*)d_in[8];
    const float* b_v   = (const float*)d_in[9];
    const float* w_o   = (const float*)d_in[10];
    const float* b_o   = (const float*)d_in[11];
    const float* relb  = (const float*)d_in[12];
    float* out = (float*)d_out;

    char* ws = (char*)d_ws;
    size_t off = 0;
    auto take = [&](size_t bytes) -> uint16_t* {
        uint16_t* p = (uint16_t*)(ws + off);
        off += (bytes + 255) & ~(size_t)255;
        return p;
    };

    const size_t XB = (size_t)MROWS * DMODEL * 2;   // 8.4 MB
    const size_t WB = (size_t)DMODEL * DMODEL * 2;  // 2.1 MB
    uint16_t* Xq = take(XB);
    uint16_t* Xk = take(XB);
    uint16_t* Xv = take(XB);
    uint16_t* Wq = take(WB);
    uint16_t* Wk = take(WB);
    uint16_t* Wv = take(WB);
    uint16_t* Wo = take(WB);
    uint16_t* Qh = take(XB);   // [B,H,S,DK]
    uint16_t* Kh = take(XB);   // [B,H,S,DK]
    uint16_t* Vt = take(XB);   // [B,H,DK,S]
    uint16_t* Ct = take(XB);   // [B,S,DMODEL]

    const int nX4 = MROWS * DMODEL / 4;
    const int nW4 = DMODEL * DMODEL / 4;
    k_cvt_bf16x4<<<nX4 / 256, 256, 0, stream>>>((const float4*)q_in, (uint2*)Xq, nX4);
    k_cvt_bf16x4<<<nX4 / 256, 256, 0, stream>>>((const float4*)k_in, (uint2*)Xk, nX4);
    k_cvt_bf16x4<<<nX4 / 256, 256, 0, stream>>>((const float4*)v_in, (uint2*)Xv, nX4);
    k_cvt_bf16x4<<<nW4 / 256, 256, 0, stream>>>((const float4*)w_q, (uint2*)Wq, nW4);
    k_cvt_bf16x4<<<nW4 / 256, 256, 0, stream>>>((const float4*)w_k, (uint2*)Wk, nW4);
    k_cvt_bf16x4<<<nW4 / 256, 256, 0, stream>>>((const float4*)w_v, (uint2*)Wv, nW4);
    k_cvt_bf16x4<<<nW4 / 256, 256, 0, stream>>>((const float4*)w_o, (uint2*)Wo, nW4);

    dim3 ggrid(DMODEL / 128, MROWS / 128);
    k_gemm_bf16<<<ggrid, 256, 0, stream>>>(Xq, Wq, b_q, Qh, nullptr, 0);
    k_gemm_bf16<<<ggrid, 256, 0, stream>>>(Xk, Wk, b_k, Kh, nullptr, 0);
    k_gemm_bf16<<<ggrid, 256, 0, stream>>>(Xv, Wv, b_v, Vt, nullptr, 1);

    dim3 agrid(S_LEN / 64, BATCH * NHEADS);
    k_attn<<<agrid, 128, 0, stream>>>(Qh, Kh, Vt, maskp, relb, Ct);

    k_gemm_bf16<<<ggrid, 256, 0, stream>>>(Ct, Wo, b_o, nullptr, out, 2);
}
